// MoDEEncoderBlock_28784870818189
// MI455X (gfx1250) — matrix operand order repaired and checked by reading, not compiled, not run on hardware
//
#include <hip/hip_runtime.h>
#include <hip/hip_bf16.h>

typedef __bf16 bf16_t;
typedef __bf16 v16bf __attribute__((ext_vector_type(16)));
typedef __bf16 v8bf  __attribute__((ext_vector_type(8)));
typedef float  v8f   __attribute__((ext_vector_type(8)));

#define N_    32
#define L_    8192
#define T_    10
#define E_    5
#define C1_   64
#define C2_   128
#define JD_   4096      // downsampled length L/2
#define LB_   128       // conv L-tile
#define JB_   64        // downsample j-tile
#define CIPD_ 136       // downsample LDS row stride (128 + 8)

__device__ __forceinline__ float mishf(float x) {
    float sp = (x > 20.0f) ? x : log1pf(expf(x));
    return x * tanhf(sp);
}

// ---------------------------------------------------------------- gates
__global__ void gate_kernel(const float* __restrict__ t, const float* __restrict__ gw,
                            const float* __restrict__ gb, float* __restrict__ g) {
    int idx = blockIdx.x * blockDim.x + threadIdx.x;
    if (idx >= N_ * C2_) return;
    int n = idx / C2_, o = idx % C2_;
    float lg[E_];
    float m = -1e30f;
    for (int e = 0; e < E_; ++e) {
        float s = gb[e * C2_ + o];
        const float* w  = gw + (size_t)(e * C2_ + o) * T_;
        const float* tv = t + (size_t)n * T_;
        #pragma unroll
        for (int k = 0; k < T_; ++k) s += tv[k] * w[k];
        lg[e] = s;
        m = fmaxf(m, s);
    }
    float sum = 0.0f;
    for (int e = 0; e < E_; ++e) { lg[e] = expf(lg[e] - m); sum += lg[e]; }
    float inv = 1.0f / sum;
    for (int e = 0; e < E_; ++e) g[((size_t)n * E_ + e) * C2_ + o] = lg[e] * inv;
}

// ------------------- per-sample mixed weights, packed in B-fragment order
// wB[((n*NS + s)*8 + ot)*512 + r*16 + e] = W(n, o = ot*16+(r&15), kappa = 32*s + 16*(e>>3) + 8*(r>>4) + (e&7))
// kappa = kk*CI + i
template<int CI>
__global__ void build_w_kernel(const float* __restrict__ g,  const float* __restrict__ k5,
                               const float* __restrict__ k3, const float* __restrict__ k1,
                               const float* __restrict__ a3, const float* __restrict__ a5,
                               bf16_t* __restrict__ wB) {
    constexpr int NS = CI * 5 / 32;
    int idx = blockIdx.x * blockDim.x + threadIdx.x;
    if (idx >= N_ * NS * 4096) return;
    int e  = idx & 15;
    int r  = (idx >> 4) & 31;
    int ot = (idx >> 9) & 7;
    int s  = (idx >> 12) % NS;
    int n  = idx / (NS * 4096);
    int o   = ot * 16 + (r & 15);
    int kap = 32 * s + ((e & 8) << 1) + (r >> 4) * 8 + (e & 7);
    int i  = kap % CI;
    int kk = kap / CI;
    const float* gn = g + (size_t)n * E_ * C2_ + o;
    size_t oi = (size_t)o * CI + i;
    float v = gn[0] * k5[oi * 5 + kk];
    if (kk >= 2) v += gn[1 * C2_] * k3[oi * 3 + (kk - 2)] + gn[3 * C2_] * a3[oi] * (1.0f / 3.0f);
    if (kk == 4) v += gn[2 * C2_] * k1[oi];
    v += gn[4 * C2_] * a5[oi] * (1.0f / 5.0f);
    wB[idx] = (bf16_t)v;
}

// downsample weights -> B-fragment order, kappa = r2*C2 + i  (K = 256, NS = 8)
__global__ void build_dwB_kernel(const float* __restrict__ dw, bf16_t* __restrict__ dwB) {
    int idx = blockIdx.x * blockDim.x + threadIdx.x;
    if (idx >= 8 * 4096) return;
    int e  = idx & 15;
    int r  = (idx >> 4) & 31;
    int ot = (idx >> 9) & 7;
    int s  = idx >> 12;
    int o   = ot * 16 + (r & 15);
    int kap = 32 * s + ((e & 8) << 1) + (r >> 4) * 8 + (e & 7);
    int i  = kap & (C2_ - 1);
    int r2 = kap >> 7;
    dwB[idx] = (bf16_t)dw[((size_t)o * C2_ + i) * 2 + r2];
}

// ---------------------------------------------------------------- conv as GEMM (WMMA)
// D[l, o] = sum_kappa Xpatch[l, kappa] * W^T[kappa, o] ; A = x-patch, B = weights.
// A per-lane 8-chunks are contiguous channels of one LDS row -> aligned ds_load_b128.
template<int CI, typename TIN>
__global__ __launch_bounds__(256) void conv_wmma_kernel(const TIN* __restrict__ xin,
                                                        const bf16_t* __restrict__ wB,
                                                        float* __restrict__ y) {
    constexpr int NS  = CI * 5 / 32;
    constexpr int CIP = CI + 8;                    // padded row stride (bank-conflict-free)
    __shared__ bf16_t tile[(LB_ + 4) * CIP];       // natural layout (row l, channel i)
    const int n     = blockIdx.y;
    const int lbase = blockIdx.x * LB_;
    const int tid   = threadIdx.x;
    const int wave  = tid >> 5;                    // = o-tile
    const int lane  = tid & 31;

    for (int idx = tid; idx < (LB_ + 4) * CI; idx += 256) {
        int i = idx % CI;
        int r = idx / CI;
        int l = lbase - 4 + r;
        float v = 0.0f;
        if (l >= 0) v = (float)xin[((size_t)n * L_ + l) * CI + i];
        tile[r * CIP + i] = (bf16_t)v;
    }
    if (lbase + LB_ < L_)
        __builtin_prefetch(&xin[((size_t)n * L_ + lbase + LB_) * CI], 0, 1);
    __syncthreads();

    v8f z = {0.f, 0.f, 0.f, 0.f, 0.f, 0.f, 0.f, 0.f};
    v8f acc[8];
    #pragma unroll
    for (int c = 0; c < 8; ++c) acc[c] = z;

    const int lane15  = lane & 15;
    const int lane_hi = lane >> 4;
    for (int s = 0; s < NS; ++s) {
        // B fragment: contiguous + wave-coalesced 32B global load
        v16bf b = *(const v16bf*)(wB + ((((size_t)n * NS + s) * 8 + wave) * 32 + lane) * 16);
        int k0  = 32 * s + lane_hi * 8;            // A chunk0 kappa base (mult of 8)
        int kk0 = k0 / CI, i0 = k0 % CI;
        int k1  = k0 + 16;
        int kk1 = k1 / CI, i1 = k1 % CI;
        #pragma unroll
        for (int c = 0; c < 8; ++c) {
            int row = c * 16 + lane15;             // l within tile
            v8bf lo = *(const v8bf*)&tile[(row + kk0) * CIP + i0];
            v8bf hi = *(const v8bf*)&tile[(row + kk1) * CIP + i1];
            v16bf a = __builtin_shufflevector(lo, hi, 0, 1, 2, 3, 4, 5, 6, 7,
                                                      8, 9, 10, 11, 12, 13, 14, 15);
            acc[c] = __builtin_amdgcn_wmma_f32_16x16x32_bf16(false, a, false, b,
                                                             (short)0, acc[c], false, false);
        }
    }

    // C: M(row l) = rr + 8*lane_hi, N(col o) = lane15 -> per-lane contiguous 8 floats
    const int o = wave * 16 + lane15;
    #pragma unroll
    for (int c = 0; c < 8; ++c) {
        size_t base = ((size_t)n * C2_ + o) * L_ + lbase + c * 16 + 8 * lane_hi;
        *(v8f*)&y[base] = acc[c];
    }
}

// ---------------------------------------------------------------- instance norm
__global__ __launch_bounds__(256) void inorm_stats_kernel(const float* __restrict__ y,
                                                          float* __restrict__ st) {
    __shared__ float sh[256], sh2[256];
    int row = blockIdx.x;                          // n*C2 + o
    const float* p = y + (size_t)row * L_;
    float s = 0.f, s2 = 0.f;
    for (int l = threadIdx.x; l < L_; l += 256) { float v = p[l]; s += v; s2 += v * v; }
    sh[threadIdx.x] = s; sh2[threadIdx.x] = s2;
    __syncthreads();
    for (int off = 128; off > 0; off >>= 1) {
        if (threadIdx.x < off) { sh[threadIdx.x] += sh[threadIdx.x + off];
                                 sh2[threadIdx.x] += sh2[threadIdx.x + off]; }
        __syncthreads();
    }
    if (threadIdx.x == 0) {
        float mu  = sh[0] * (1.0f / L_);
        float var = sh2[0] * (1.0f / L_) - mu * mu;
        st[row * 2]     = mu;
        st[row * 2 + 1] = rsqrtf(var + 1e-5f);
    }
}

__global__ void inorm_apply1_kernel(const float* __restrict__ y, const float* __restrict__ st,
                                    const float* __restrict__ gamma, const float* __restrict__ beta,
                                    bf16_t* __restrict__ h) {
    size_t idx = (size_t)blockIdx.x * 256 + threadIdx.x;
    if (idx >= (size_t)N_ * L_ * C2_) return;
    int o = (int)(idx % C2_);
    size_t nl = idx / C2_;
    int l = (int)(nl % L_);
    int n = (int)(nl / L_);
    int row = n * C2_ + o;
    float v = y[(size_t)row * L_ + l];
    v = (v - st[row * 2]) * st[row * 2 + 1] * gamma[o] + beta[o];
    h[idx] = (bf16_t)mishf(v);
}

__global__ void inorm_apply2_kernel(const float* __restrict__ y, const float* __restrict__ st,
                                    const float* __restrict__ gamma, const float* __restrict__ beta,
                                    float* __restrict__ skip, bf16_t* __restrict__ xsb) {
    size_t idx = (size_t)blockIdx.x * 256 + threadIdx.x;
    if (idx >= (size_t)N_ * L_ * C2_) return;
    int o = (int)(idx % C2_);
    size_t nl = idx / C2_;
    int l = (int)(nl % L_);
    int n = (int)(nl / L_);
    int row = n * C2_ + o;
    float v = y[(size_t)row * L_ + l];
    v = (v - st[row * 2]) * st[row * 2 + 1] * gamma[o] + beta[o];
    float m = mishf(v);
    skip[idx] = m;
    xsb[idx]  = (bf16_t)m;
}

// ---------------------------------------------------------------- downsample (WMMA)
// D[j, o] = sum_kappa A[j, kappa] * dwB[kappa, o], kappa = r2*128 + i, K = 256
__global__ __launch_bounds__(256) void down_wmma_kernel(const bf16_t* __restrict__ xs,
                                                        const bf16_t* __restrict__ dwB,
                                                        float* __restrict__ xdpre) {
    __shared__ bf16_t tile[2 * JB_ * CIPD_];       // rows l = 2*jbase .. +127
    const int n     = blockIdx.y;
    const int jbase = blockIdx.x * JB_;
    const int tid   = threadIdx.x;
    const int wave  = tid >> 5;
    const int lane  = tid & 31;

    for (int idx = tid; idx < 2 * JB_ * C2_; idx += 256) {
        int i = idx % C2_;
        int r = idx / C2_;
        tile[r * CIPD_ + i] = xs[((size_t)n * L_ + 2 * jbase + r) * C2_ + i];
    }
    __syncthreads();

    v8f z = {0.f, 0.f, 0.f, 0.f, 0.f, 0.f, 0.f, 0.f};
    v8f acc[4];
    #pragma unroll
    for (int c = 0; c < 4; ++c) acc[c] = z;

    const int lane15  = lane & 15;
    const int lane_hi = lane >> 4;
    #pragma unroll
    for (int s = 0; s < 8; ++s) {
        v16bf b = *(const v16bf*)(dwB + (((size_t)s * 8 + wave) * 32 + lane) * 16);
        int k0 = 32 * s + lane_hi * 8;
        int r0 = k0 >> 7, i0 = k0 & (C2_ - 1);
        int k1 = k0 + 16;
        int r1 = k1 >> 7, i1 = k1 & (C2_ - 1);
        #pragma unroll
        for (int c = 0; c < 4; ++c) {
            int j_loc = c * 16 + lane15;
            v8bf lo = *(const v8bf*)&tile[(2 * j_loc + r0) * CIPD_ + i0];
            v8bf hi = *(const v8bf*)&tile[(2 * j_loc + r1) * CIPD_ + i1];
            v16bf a = __builtin_shufflevector(lo, hi, 0, 1, 2, 3, 4, 5, 6, 7,
                                                      8, 9, 10, 11, 12, 13, 14, 15);
            acc[c] = __builtin_amdgcn_wmma_f32_16x16x32_bf16(false, a, false, b,
                                                             (short)0, acc[c], false, false);
        }
    }

    const int o = wave * 16 + lane15;
    #pragma unroll
    for (int c = 0; c < 4; ++c) {
        size_t base = ((size_t)n * C2_ + o) * JD_ + jbase + c * 16 + 8 * lane_hi;
        *(v8f*)&xdpre[base] = acc[c];
    }
}

// ---------------------------------------------------------------- batch norm
__global__ __launch_bounds__(256) void bn_stats_kernel(const float* __restrict__ xd,
                                                       float* __restrict__ st) {
    __shared__ float sh[256], sh2[256];
    int o = blockIdx.x;
    float s = 0.f, s2 = 0.f;
    for (int r = threadIdx.x; r < N_ * JD_; r += 256) {
        int n = r / JD_, j = r % JD_;
        float v = xd[((size_t)n * C2_ + o) * JD_ + j];
        s += v; s2 += v * v;
    }
    sh[threadIdx.x] = s; sh2[threadIdx.x] = s2;
    __syncthreads();
    for (int off = 128; off > 0; off >>= 1) {
        if (threadIdx.x < off) { sh[threadIdx.x] += sh[threadIdx.x + off];
                                 sh2[threadIdx.x] += sh2[threadIdx.x + off]; }
        __syncthreads();
    }
    if (threadIdx.x == 0) {
        const float invc = 1.0f / (float)(N_ * JD_);
        float mu  = sh[0] * invc;
        float var = sh2[0] * invc - mu * mu;
        st[o * 2]     = mu;
        st[o * 2 + 1] = rsqrtf(var + 1e-5f);
    }
}

__global__ void bn_apply_kernel(const float* __restrict__ xd, const float* __restrict__ st,
                                const float* __restrict__ dg, const float* __restrict__ db,
                                float* __restrict__ out) {
    size_t idx = (size_t)blockIdx.x * 256 + threadIdx.x;
    if (idx >= (size_t)N_ * C2_ * JD_) return;
    int o = (int)((idx / JD_) % C2_);
    float v = xd[idx];
    v = (v - st[o * 2]) * st[o * 2 + 1] * dg[o] + db[o];
    out[idx] = mishf(v);
}

// ---------------------------------------------------------------- launch
extern "C" void kernel_launch(void* const* d_in, const int* in_sizes, int n_in,
                              void* d_out, int out_size, void* d_ws, size_t ws_size,
                              hipStream_t stream) {
    const float* x        = (const float*)d_in[0];
    const float* t        = (const float*)d_in[1];
    const float* l1_k5    = (const float*)d_in[2];
    const float* l1_k3    = (const float*)d_in[3];
    const float* l1_k1    = (const float*)d_in[4];
    const float* l1_a3    = (const float*)d_in[5];
    const float* l1_a5    = (const float*)d_in[6];
    const float* l1_gw    = (const float*)d_in[7];
    const float* l1_gb    = (const float*)d_in[8];
    const float* l1_gamma = (const float*)d_in[9];
    const float* l1_beta  = (const float*)d_in[10];
    const float* l2_k5    = (const float*)d_in[11];
    const float* l2_k3    = (const float*)d_in[12];
    const float* l2_k1    = (const float*)d_in[13];
    const float* l2_a3    = (const float*)d_in[14];
    const float* l2_a5    = (const float*)d_in[15];
    const float* l2_gw    = (const float*)d_in[16];
    const float* l2_gb    = (const float*)d_in[17];
    const float* l2_gamma = (const float*)d_in[18];
    const float* l2_beta  = (const float*)d_in[19];
    const float* dw       = (const float*)d_in[20];
    const float* dg       = (const float*)d_in[21];
    const float* db       = (const float*)d_in[22];

    float* out_xd   = (float*)d_out;                         // (N, C2, JD)
    float* out_skip = out_xd + (size_t)N_ * C2_ * JD_;       // (N, L, C2)

    char* ws = (char*)d_ws;
    float*  g1    = (float*)(ws + 0x000000);                 // 80 KB
    float*  g2    = (float*)(ws + 0x020000);
    bf16_t* wB1   = (bf16_t*)(ws + 0x040000);                // 2.5 MB
    bf16_t* wB2   = (bf16_t*)(ws + 0x2C0000);                // 5 MB
    bf16_t* dwB   = (bf16_t*)(ws + 0x7C0000);                // 64 KB
    float*  st1   = (float*)(ws + 0x7D0000);
    float*  st2   = (float*)(ws + 0x7D8000);
    float*  bnst  = (float*)(ws + 0x7E0000);
    float*  ybuf  = (float*)(ws + 0x800000);                 // 128 MB (y1 -> y2 -> xd_pre)
    bf16_t* bbuf  = (bf16_t*)(ws + 0x800000 + (size_t)N_ * C2_ * L_ * 4); // 64 MB (h -> xs)
    float*  xdpre = ybuf;                                    // alias: y dead by then

    const size_t NLC = (size_t)N_ * L_ * C2_;

    gate_kernel<<<(N_ * C2_ + 255) / 256, 256, 0, stream>>>(t, l1_gw, l1_gb, g1);
    gate_kernel<<<(N_ * C2_ + 255) / 256, 256, 0, stream>>>(t, l2_gw, l2_gb, g2);

    build_w_kernel<C1_><<<(N_ * 10 * 4096 + 255) / 256, 256, 0, stream>>>(
        g1, l1_k5, l1_k3, l1_k1, l1_a3, l1_a5, wB1);
    build_w_kernel<C2_><<<(N_ * 20 * 4096 + 255) / 256, 256, 0, stream>>>(
        g2, l2_k5, l2_k3, l2_k1, l2_a3, l2_a5, wB2);
    build_dwB_kernel<<<(8 * 4096 + 255) / 256, 256, 0, stream>>>(dw, dwB);

    conv_wmma_kernel<C1_, float><<<dim3(L_ / LB_, N_), 256, 0, stream>>>(x, wB1, ybuf);
    inorm_stats_kernel<<<N_ * C2_, 256, 0, stream>>>(ybuf, st1);
    inorm_apply1_kernel<<<(int)((NLC + 255) / 256), 256, 0, stream>>>(
        ybuf, st1, l1_gamma, l1_beta, bbuf);

    conv_wmma_kernel<C2_, bf16_t><<<dim3(L_ / LB_, N_), 256, 0, stream>>>(bbuf, wB2, ybuf);
    inorm_stats_kernel<<<N_ * C2_, 256, 0, stream>>>(ybuf, st2);
    inorm_apply2_kernel<<<(int)((NLC + 255) / 256), 256, 0, stream>>>(
        ybuf, st2, l2_gamma, l2_beta, out_skip, bbuf);

    down_wmma_kernel<<<dim3(JD_ / JB_, N_), 256, 0, stream>>>(bbuf, dwB, xdpre);
    bn_stats_kernel<<<C2_, 256, 0, stream>>>(xdpre, bnst);
    bn_apply_kernel<<<(int)(((size_t)N_ * C2_ * JD_ + 255) / 256), 256, 0, stream>>>(
        xdpre, bnst, dg, db, out_xd);
}
